// BlstmCell_36644660969530
// MI455X (gfx1250) — compile-verified, matrix operand order checked by reading, not run on hardware
//
#include <hip/hip_runtime.h>

// ---------------------------------------------------------------- constants
#define SEQ   512
#define BATCH 128
#define IN    256
#define HID   256
#define CS    512   // HID + IN
#define NDIR  2
#define NGATE 4

// workspace byte offsets (256B aligned)
#define CTR_OFF   0u                     // 1024 B barrier counters
#define XBF_OFF   1024ull                // SEQ*BATCH*IN bf16 = 32 MB
#define PBX_OFF   (XBF_OFF + 33554432ull)    // packed Wx  (1 MB)
#define PBH_OFF   (PBX_OFF + 1048576ull)     // packed Wh  (1 MB)
#define BIAS_OFF  (PBH_OFF + 1048576ull)     // 2048 f32   (8 KB)
#define HBUF_OFF  (BIAS_OFF + 8192ull)       // 2 bufs * 2 dirs * 128*256 bf16 = 256 KB
#define GPRE_OFF  (HBUF_OFF + 262144ull)     // packed D-frags: 2*512*4*8*16 tiles * 512B = 256 MB

typedef __attribute__((ext_vector_type(16))) __bf16        v16bf;
typedef __attribute__((ext_vector_type(8)))  float         v8f;
typedef __attribute__((ext_vector_type(4)))  unsigned int  v4u;
typedef __attribute__((ext_vector_type(2)))  unsigned int  v2u;

union FragB {
    v16bf          bf;
    v4u            u[2];
    unsigned short s[16];
};
union FragH {               // half fragment: 8 bf16 (one b128)
    v4u            u;
    unsigned short s[8];
};

struct WPtrs { const float* w[8]; const float* b[8]; };

static __device__ __forceinline__ unsigned short f2bf(float f) {
    unsigned u = __float_as_uint(f);
    u += 0x7FFFu + ((u >> 16) & 1u);          // round-to-nearest-even
    return (unsigned short)(u >> 16);
}
static __device__ __forceinline__ float bf2f(unsigned short s) {
    return __uint_as_float(((unsigned)s) << 16);
}
static __device__ __forceinline__ float sigmoidf(float x) {
    return 1.0f / (1.0f + __expf(-x));
}

// ------------------------------------------------------------ X -> bf16
__global__ void k_convert_x(const float* __restrict__ X,
                            unsigned short* __restrict__ XBF) {
    long long i = ((long long)blockIdx.x * 256 + threadIdx.x) * 4;
    if (i >= (long long)SEQ * BATCH * IN) return;
    float4 v = *(const float4*)(X + i);
    v2u r;
    r[0] = (unsigned)f2bf(v.x) | ((unsigned)f2bf(v.y) << 16);
    r[1] = (unsigned)f2bf(v.z) | ((unsigned)f2bf(v.w) << 16);
    *(v2u*)(XBF + i) = r;
}

// ---------------------------------------- pack W into WMMA-B fragment tiles
// B fragment layout (16x16x32 bf16): lane l holds column N = n0+(l&15);
// element e (v = e>>1, hi = e&1):
//   K = kt*32 + ((l>=16)?8:0) + (v<4 ? 0 : 16) + (v&3)*2 + hi
__global__ void k_pack_w(WPtrs p,
                         unsigned short* __restrict__ PBX,
                         unsigned short* __restrict__ PBH,
                         float* __restrict__ BIAS) {
    int gid  = blockIdx.x * 256 + threadIdx.x;   // 0..65535
    int lane = gid & 31;
    int tile = gid >> 5;                          // 0..2047
    int khalf = (lane >> 4) ? 8 : 0;

    FragB f;
    if (tile < 1024) {                            // Wx: 128 ntiles x 8 ktiles
        int ntile = tile >> 3, kt = tile & 7;
        int N = ntile * 16 + (lane & 15);
        int dirg = N >> 8, j = N & 255;           // dirg = dir*4+gate
        const float* W = p.w[dirg];
        #pragma unroll
        for (int e = 0; e < 16; ++e) {
            int v = e >> 1, hi = e & 1;
            int K = kt * 32 + khalf + ((v < 4) ? 0 : 16) + (v & 3) * 2 + hi;
            f.s[e] = f2bf(W[(size_t)j * CS + HID + K]);   // x-part of W
        }
        unsigned short* dst = PBX + (size_t)tile * 512 + lane * 16;
        *(v4u*)dst       = f.u[0];
        *(v4u*)(dst + 8) = f.u[1];
    } else {                                      // Wh: per dir 64 ntiles x 8 ktiles
        int t2  = tile - 1024;
        int dir = t2 >> 9;
        int t3  = t2 & 511;
        int nt  = t3 >> 3, kt = t3 & 7;
        int njc = nt * 16 + (lane & 15);          // gate-major column index
        int g = njc >> 8, j = njc & 255;
        const float* W = p.w[dir * 4 + g];
        #pragma unroll
        for (int e = 0; e < 16; ++e) {
            int v = e >> 1, hi = e & 1;
            int K = kt * 32 + khalf + ((v < 4) ? 0 : 16) + (v & 3) * 2 + hi;
            f.s[e] = f2bf(W[(size_t)j * CS + K]);         // h-part of W
        }
        unsigned short* dst = PBH + (size_t)dir * 262144 + (size_t)t3 * 512 + lane * 16;
        *(v4u*)dst       = f.u[0];
        *(v4u*)(dst + 8) = f.u[1];
    }
    if (gid < 2048) {                             // bias: BIAS[dirg*256+j]
        int dirg = gid >> 8, j = gid & 255;
        BIAS[gid] = p.b[dirg][j];
    }
}

// -------------------------------------------- GEMM1: input projections
// Gates[m][n] = sum_k Xbf[m][k]*Wx[n][k] + bias[n], m=(t,b), n=(dir,gate,j)
// wave computes 16 rows x 128 cols (8 N tiles), K=256 in 8 steps of 32.
// Output written as PACKED D-fragments:
//   GPRE tile index = (((dir*SEQ+t)*4+g)*8 + mtB)*16 + nt,  tile = 32 lanes x 8 bf16
__global__ void __launch_bounds__(256, 2)
k_gemm_in(const unsigned short* __restrict__ XBF,
          const unsigned short* __restrict__ PBX,
          const float* __restrict__ BIAS,
          unsigned short* __restrict__ GPRE) {
    int wid  = threadIdx.x >> 5, lane = threadIdx.x & 31;
    int gid  = blockIdx.x * 8 + wid;              // 0..65535
    int mt   = gid >> 4, cg = gid & 15;
    int m0   = mt << 4;
    int rowA = m0 + (lane & 15);
    int khalf = (lane >> 4) << 3;

    v8f acc[8];
    #pragma unroll
    for (int nn = 0; nn < 8; ++nn) acc[nn] = (v8f)0.0f;

    for (int kt = 0; kt < 8; ++kt) {
        FragB A;
        const unsigned short* ap = XBF + (size_t)rowA * IN + kt * 32 + khalf;
        A.u[0] = *(const v4u*)ap;
        A.u[1] = *(const v4u*)(ap + 16);
        // software-pipeline B fragments (load nn+1 before consuming nn)
        const unsigned short* bbase = PBX + ((size_t)cg * 8 * 8 + kt) * 512 + lane * 16;
        FragB Bc, Bn;
        Bc.u[0] = *(const v4u*)bbase;
        Bc.u[1] = *(const v4u*)(bbase + 8);
        #pragma unroll
        for (int nn = 0; nn < 8; ++nn) {
            if (nn < 7) {
                const unsigned short* bp = bbase + (size_t)(nn + 1) * 8 * 512;
                Bn.u[0] = *(const v4u*)bp;
                Bn.u[1] = *(const v4u*)(bp + 8);
            }
            acc[nn] = __builtin_amdgcn_wmma_f32_16x16x32_bf16(
                false, A.bf, false, Bc.bf, (short)0, acc[nn], false, false);
            Bc = Bn;
        }
    }

    // epilogue: +bias, f32->bf16, one b128 store per N tile (packed frag)
    int t   = m0 >> 7;
    int mtB = (m0 & 127) >> 4;                    // batch tile 0..7
    #pragma unroll
    for (int nn = 0; nn < 8; ++nn) {
        int ntg  = cg * 8 + nn;                   // global ntile 0..127
        int dirg = ntg >> 4, nt = ntg & 15;
        int col  = ntg * 16 + (lane & 15);
        float bias = BIAS[col];
        v4u o;
        #pragma unroll
        for (int i = 0; i < 4; ++i) {
            o[i] = (unsigned)f2bf(acc[nn][2 * i] + bias)
                 | ((unsigned)f2bf(acc[nn][2 * i + 1] + bias) << 16);
        }
        size_t tile = ((((size_t)(dirg >> 2) * SEQ + t) * NGATE + (dirg & 3)) * 8
                       + mtB) * 16 + nt;
        *(v4u*)(GPRE + tile * 256 + lane * 8) = o;
    }
}

// ------------------------------------------------- inter-WG step barrier
static __device__ __forceinline__ void gbar(unsigned int* ctr, unsigned int target) {
    __syncthreads();
    if (threadIdx.x == 0) {
        __threadfence();                      // release h stores
        atomicAdd(ctr, 1u);
        while (__atomic_load_n(ctr, __ATOMIC_RELAXED) < target) {
            __builtin_amdgcn_s_sleep(2);
        }
    }
    __syncthreads();
    __threadfence();                          // acquire peers' h stores
}

// -------------------------------------------- persistent recurrence kernel
// grid = 16 WGs: dir = blk>>3, slice = blk&7 owns cols [slice*32, slice*32+32)
// of all 4 gates. Wh slice (64 KB) lives in LDS for all 512 steps.
__global__ void __launch_bounds__(256, 1)
k_recur(const unsigned short* __restrict__ GPRE,
        const unsigned short* __restrict__ PBH,
        unsigned short* __restrict__ HBUF,
        float* __restrict__ OUT,
        unsigned int* __restrict__ CTR) {
    __shared__ unsigned int ldsW[16384];      // 64 KB: 64 tiles x 256 u32
    const int dir   = blockIdx.x >> 3;
    const int slice = blockIdx.x & 7;
    const int j0    = slice * 32;
    const int tid   = threadIdx.x;
    const int wid   = tid >> 5, lane = tid & 31;

    // stage recurrent-weight fragments into LDS via async global->LDS copies
    // (ASYNCcnt-tracked; per-lane 16B), reshuffling tiles into local order.
    {
        unsigned ldsbase =
            (unsigned)(uintptr_t)(__attribute__((address_space(3))) unsigned int*)ldsW;
        const unsigned short* srcdir = PBH + (size_t)dir * 262144;
        #pragma unroll
        for (int it = 0; it < 16; ++it) {
            int i    = it * 256 + tid;            // 4096 chunk-ops of 16B
            int tile = i >> 6, chunk = i & 63;    // local tile, 16B chunk
            int nt = tile >> 3, kt = tile & 7;
            int g = nt >> 1, jt = nt & 1;
            int gnt = g * 16 + (j0 >> 4) + jt;    // global ntile within dir
            unsigned long long gaddr = (unsigned long long)(uintptr_t)
                (srcdir + ((size_t)(gnt * 8 + kt)) * 512 + chunk * 8);
            unsigned laddr = ldsbase + (unsigned)(tile * 1024 + chunk * 16);
            asm volatile("global_load_async_to_lds_b128 %0, %1, off"
                         :: "v"(laddr), "v"(gaddr) : "memory");
        }
        asm volatile("s_wait_asynccnt 0x0" ::: "memory");
    }
    // zero h double-buffer 0 (own column slice)
    for (int i = tid; i < 2048; i += 256) {
        int b = i >> 4, jc = i & 15;
        ((unsigned int*)HBUF)[((size_t)dir * BATCH + b) * 128 + (j0 >> 1) + jc] = 0u;
    }
    float c[2][8];
    #pragma unroll
    for (int jt = 0; jt < 2; ++jt)
        #pragma unroll
        for (int r = 0; r < 8; ++r) c[jt][r] = 0.0f;

    gbar(&CTR[dir], 8u);                      // init barrier (also covers LDS fill)

    const int rowb = wid * 16 + ((lane >> 4) << 3);
    for (int s = 0; s < SEQ; ++s) {
        int t = dir ? (SEQ - 1 - s) : s;
        const unsigned short* hprev =
            HBUF + ((size_t)((s & 1) * NDIR + dir) * BATCH) * HID;
        unsigned short* hnext =
            HBUF + ((size_t)(((s + 1) & 1) * NDIR + dir) * BATCH) * HID;

        // A fragments: h_prev rows for this wave, all 8 K tiles
        FragB A[8];
        {
            int row = wid * 16 + (lane & 15);
            int khalf = (lane >> 4) << 3;
            const unsigned short* ap = hprev + (size_t)row * HID + khalf;
            #pragma unroll
            for (int kt = 0; kt < 8; ++kt) {
                A[kt].u[0] = *(const v4u*)(ap + kt * 32);
                A[kt].u[1] = *(const v4u*)(ap + kt * 32 + 16);
            }
        }

        const size_t ttile = (((size_t)dir * SEQ + t) * NGATE);
        #pragma unroll
        for (int jt = 0; jt < 2; ++jt) {
            int colj = j0 + jt * 16 + (lane & 15);
            v8f acc[4];
            #pragma unroll
            for (int g = 0; g < 4; ++g) {
                // seed: packed D-fragment, single b128 load
                size_t tile = ((ttile + g) * 8 + wid) * 16 + (j0 >> 4) + jt;
                FragH seed;
                seed.u = *(const v4u*)(GPRE + tile * 256 + lane * 8);
                v8f a;
                #pragma unroll
                for (int r = 0; r < 8; ++r) a[r] = bf2f(seed.s[r]);
                #pragma unroll
                for (int kt = 0; kt < 8; ++kt) {
                    const unsigned int* bp =
                        &ldsW[((g * 2 + jt) * 8 + kt) * 256 + lane * 8];
                    FragB B;
                    B.u[0] = *(const v4u*)bp;
                    B.u[1] = *(const v4u*)(bp + 4);
                    a = __builtin_amdgcn_wmma_f32_16x16x32_bf16(
                        false, A[kt].bf, false, B.bf, (short)0, a, false, false);
                }
                acc[g] = a;
            }
            // LSTM pointwise on D-fragment layout
            #pragma unroll
            for (int r = 0; r < 8; ++r) {
                float fg = sigmoidf(acc[0][r]);
                float ig = sigmoidf(acc[1][r]);
                float cg = tanhf(acc[2][r]);
                float og = sigmoidf(acc[3][r]);
                float cn = fg * c[jt][r] + ig * cg;
                c[jt][r] = cn;
                float h = og * tanhf(cn);
                int b = rowb + r;
                hnext[(size_t)b * HID + colj] = f2bf(h);
                OUT[((size_t)t * BATCH + b) * (2 * HID) + dir * HID + colj] = h;
            }
        }

        // prefetch next timestep's own 32KB slice: one 128B line per thread
        if (s < SEQ - 1) {
            int t2 = dir ? (SEQ - 2 - s) : (s + 1);
            const unsigned short* pf =
                GPRE + (((size_t)dir * SEQ + t2) * NGATE) * 8 * 16 * 256;
            int g = tid >> 6, mtp = (tid >> 3) & 7;
            int ntl = (j0 >> 4) + ((tid >> 2) & 1), q = tid & 3;
            size_t tile = ((size_t)g * 8 + mtp) * 16 + ntl;
            __builtin_prefetch((const char*)(pf + tile * 256) + q * 128, 0, 1);
        }
        gbar(&CTR[dir], 8u * (unsigned)(s + 2));
    }
}

// ------------------------------------------------------------------ launch
extern "C" void kernel_launch(void* const* d_in, const int* in_sizes, int n_in,
                              void* d_out, int out_size, void* d_ws, size_t ws_size,
                              hipStream_t stream) {
    (void)in_sizes; (void)n_in; (void)out_size; (void)ws_size;
    const float* X = (const float*)d_in[0];
    WPtrs p;
    for (int d = 0; d < 2; ++d)
        for (int g = 0; g < 4; ++g) {
            p.w[d * 4 + g] = (const float*)d_in[1 + d * 8 + g * 2];
            p.b[d * 4 + g] = (const float*)d_in[2 + d * 8 + g * 2];
        }
    char* ws = (char*)d_ws;
    unsigned int*   CTR  = (unsigned int*)(ws + CTR_OFF);
    unsigned short* XBF  = (unsigned short*)(ws + XBF_OFF);
    unsigned short* PBX  = (unsigned short*)(ws + PBX_OFF);
    unsigned short* PBH  = (unsigned short*)(ws + PBH_OFF);
    float*          BIAS = (float*)(ws + BIAS_OFF);
    unsigned short* HBUF = (unsigned short*)(ws + HBUF_OFF);
    unsigned short* GPRE = (unsigned short*)(ws + GPRE_OFF);

    hipMemsetAsync(ws, 0, 1024, stream);                       // barrier counters
    k_convert_x<<<16384, 256, 0, stream>>>(X, XBF);
    k_pack_w   <<<256,   256, 0, stream>>>(p, PBX, PBH, BIAS);
    k_gemm_in  <<<8192,  256, 0, stream>>>(XBF, PBX, BIAS, GPRE);
    k_recur    <<<16,    256, 0, stream>>>(GPRE, PBH, HBUF, (float*)d_out, CTR);
}
